// SVConvolution_52252572123499
// MI455X (gfx1250) — matrix operand order repaired and checked by reading, not compiled, not run on hardware
//
#include <hip/hip_runtime.h>
#include <hip/hip_bf16.h>

// CDNA5 / gfx1250 spatially-varying depthwise full-correlation via WMMA.
//
// Per (image n, region p):
//   Xpad[64][190]  = zero-padded, 2x nearest-upsampled 32x32 block (f16, LDS)
//   For v in [0,127):   C_v[ki,r] = sum_kj PSF[ki,kj] * Xpad[r, v+kj]   (64x64x64 GEMM, WMMA f16->f32)
//   out[u,v] += C_v[ki,r] where u = r - ki + 63   (ds_add_f32 into LDS accumulator)
//   Flush 127x127 accumulator with global f32 atomics (overlapping region splat).
//
// Main loop processes TWO output columns per iteration: two independent
// WMMA accumulation chains interleave so each chain's co-exec hazard slots
// (f16 WMMA->WMMA RAW: 1 NOP + 4 coexec, ISA 7.12.1) are filled with real work.

typedef __attribute__((ext_vector_type(16))) _Float16 v16h;
typedef __attribute__((ext_vector_type(8)))  float    v8f;

#define IMG      512
#define BLOCK    32
#define PSFN     64
#define P_REG    256
#define SENS     1024
#define OUTK     127                 // 2*64 - 1
#define XPAD_W   190                 // 64 + 2*63 ; 95 dwords/row (odd) -> bank-conflict-free column reads
#define S_OUT_F  (OUTK * OUTK)       // 16129 floats
#define S_OUT_B  (S_OUT_F * 4)       // 64516 bytes (4-aligned)
#define S_X_H    (PSFN * XPAD_W)     // 12160 halfs
#define SMEM_B   (S_OUT_B + S_X_H * 2)  // 88836 bytes < 320KB/WGP

__global__ void sv_zero_kernel(float* __restrict__ out, int n) {
    int i = blockIdx.x * blockDim.x + threadIdx.x;
    if (i < n) out[i] = 0.0f;
}

__global__ void __launch_bounds__(512)
sv_conv_kernel(const float* __restrict__ imgs,
               const float* __restrict__ psf,
               const float* __restrict__ xcent,
               const float* __restrict__ ycent,
               float* __restrict__ out)
{
    extern __shared__ char smem[];
    float*    s_out = (float*)smem;                    // 127*127 f32 accumulator
    _Float16* s_x   = (_Float16*)(smem + S_OUT_B);     // 64 x 190 f16 padded rows

    const int tid = threadIdx.x;
    const int n   = blockIdx.x >> 8;    // image index
    const int p   = blockIdx.x & 255;   // region index
    const int bh  = p & 15;             // H-block (rows = p % 16 in reference)
    const int bw  = p >> 4;             // W-block

    // ---- zero LDS accumulator ----
    for (int i = tid; i < S_OUT_F; i += 512) s_out[i] = 0.0f;

    // ---- build padded, 2x nearest-upsampled block in LDS (f32 -> f16) ----
    const float* ib = imgs + ((size_t)n * IMG + (size_t)bh * BLOCK) * IMG + (size_t)bw * BLOCK;
    for (int idx = tid; idx < S_X_H; idx += 512) {
        int i  = idx / XPAD_W;
        int j  = idx - i * XPAD_W;
        int jj = j - 63;                                 // full-conv padding K-1 = 63
        float val = 0.0f;
        if (jj >= 0 && jj < 64) val = ib[(i >> 1) * IMG + (jj >> 1)];  // nearest upsample x2
        s_x[i * XPAD_W + j] = (_Float16)val;
    }
    __syncthreads();

    // ---- per-wave 16x16 tile of C[ki, r] ----
    const int wave = tid >> 5;
    const int lane = tid & 31;
    const int it   = wave & 3;       // ki tile (GEMM M)
    const int rt   = wave >> 2;      // r  tile (GEMM N)
    const int l15  = lane & 15;
    const int hi   = lane >> 4;      // lane half (0: lanes 0-15, 1: lanes 16-31)

    // A = PSF fragments, fixed across the whole v loop.
    // A k order used here (contiguous, matching B): lanes0-15 K=0..15, lanes16-31 K=16..31
    v16h a0, a1;
    {
        const float* wrow = psf + ((size_t)p * PSFN + (size_t)(it * 16 + l15)) * PSFN;
        const int klo8 = hi * 8;
#pragma unroll
        for (int e = 0; e < 16; ++e) {
            int kj = 2 * klo8 + e;
            a0[e] = (_Float16)wrow[kj];        // K-chunk 0: kj in [0,32)
            a1[e] = (_Float16)wrow[32 + kj];   // K-chunk 1: kj in [32,64)
        }
    }

    // B = sliding X window. B 32x16 operand, k order used here (same as A): col N = lane%16,
    // lanes0-15 hold K=0..15 sequential, lanes16-31 hold K=16..31.
    const _Float16* xrow = s_x + (size_t)(rt * 16 + l15) * XPAD_W + hi * 16;
    const int rbase  = rt * 16 + l15;            // r  (N of tile)
    const int kibase = it * 16 + hi * 8;         // ki base (M of tile, +g per accum reg)
    const int ubase  = rbase - kibase + 63;      // u for g = 0

    // ---- main loop: two output columns per iteration (independent WMMA chains) ----
#pragma unroll 1
    for (int v = 0; v + 1 < OUTK; v += 2) {
        v16h b0a, b1a, b0b, b1b;
#pragma unroll
        for (int e = 0; e < 16; ++e) {
            b0a[e] = xrow[v + e];            // col v,   kj-chunk 0
            b1a[e] = xrow[v + 32 + e];       // col v,   kj-chunk 1
            b0b[e] = xrow[v + 1 + e];        // col v+1, kj-chunk 0
            b1b[e] = xrow[v + 33 + e];       // col v+1, kj-chunk 1
        }
        v8f ca = {0.f, 0.f, 0.f, 0.f, 0.f, 0.f, 0.f, 0.f};
        v8f cb = {0.f, 0.f, 0.f, 0.f, 0.f, 0.f, 0.f, 0.f};
        ca = __builtin_amdgcn_wmma_f32_16x16x32_f16(false, a0, false, b0a, (short)0, ca, false, false);
        cb = __builtin_amdgcn_wmma_f32_16x16x32_f16(false, a0, false, b0b, (short)0, cb, false, false);
        ca = __builtin_amdgcn_wmma_f32_16x16x32_f16(false, a1, false, b1a, (short)0, ca, false, false);
        cb = __builtin_amdgcn_wmma_f32_16x16x32_f16(false, a1, false, b1b, (short)0, cb, false, false);

        // C/D layout: VGPR g -> M = g + 8*hi (= ki local), N = lane%16 (= r local).
        // Diagonal scatter: u = r - ki + 63, always in [0,127).
#pragma unroll
        for (int g = 0; g < 8; ++g) {
            atomicAdd(&s_out[(ubase - g) * OUTK + v],     ca[g]);   // ds_add_f32
            atomicAdd(&s_out[(ubase - g) * OUTK + v + 1], cb[g]);
        }
    }
    {   // ---- remainder column v = 126 ----
        const int v = OUTK - 1;
        v16h b0, b1;
#pragma unroll
        for (int e = 0; e < 16; ++e) {
            b0[e] = xrow[v + e];
            b1[e] = xrow[v + 32 + e];
        }
        v8f c = {0.f, 0.f, 0.f, 0.f, 0.f, 0.f, 0.f, 0.f};
        c = __builtin_amdgcn_wmma_f32_16x16x32_f16(false, a0, false, b0, (short)0, c, false, false);
        c = __builtin_amdgcn_wmma_f32_16x16x32_f16(false, a1, false, b1, (short)0, c, false, false);
#pragma unroll
        for (int g = 0; g < 8; ++g)
            atomicAdd(&s_out[(ubase - g) * OUTK + v], c[g]);
    }
    __syncthreads();

    // ---- splat accumulated 127x127 patch into sensor (overlap -> global atomics) ----
    const int ic = 512 + __float2int_rn(xcent[p] * 1.0e6f);   // round(x_center / PIX_SIZE)
    const int jc = 512 + __float2int_rn(ycent[p] * 1.0e6f);
    float* ob = out + (size_t)n * SENS * SENS;
    for (int idx = tid; idx < S_OUT_F; idx += 512) {
        int u = idx / OUTK;
        int v = idx - u * OUTK;
        int I = ic - 63 + u;
        int J = jc - 63 + v;
        if ((unsigned)I < (unsigned)SENS && (unsigned)J < (unsigned)SENS)
            atomicAdd(&ob[(size_t)I * SENS + J], s_out[idx]);
    }
}

extern "C" void kernel_launch(void* const* d_in, const int* in_sizes, int n_in,
                              void* d_out, int out_size, void* d_ws, size_t ws_size,
                              hipStream_t stream) {
    const float* imgs  = (const float*)d_in[0];   // [N,512,512] f32
    const float* psf   = (const float*)d_in[1];   // [256,64,64] f32
    // d_in[2], d_in[3] = X, Y meshgrids (only shapes matter; SENS=1024 fixed)
    const float* xcent = (const float*)d_in[4];   // [256] f32
    const float* ycent = (const float*)d_in[5];   // [256] f32
    float* out = (float*)d_out;                   // [N,1024,1024] f32

    const int nImg = in_sizes[0] / (IMG * IMG);   // 8

    sv_zero_kernel<<<(out_size + 255) / 256, 256, 0, stream>>>(out, out_size);
    sv_conv_kernel<<<dim3(nImg * P_REG), dim3(512), SMEM_B, stream>>>(imgs, psf, xcent, ycent, out);
}
